// myLSTM_70884140253930
// MI455X (gfx1250) — matrix operand order repaired and checked by reading, not compiled, not run on hardware
//
#include <hip/hip_runtime.h>
#include <hip/hip_bf16.h>
#include <stdint.h>

// Problem constants (match reference)
#define B_  64
#define T_  512
#define D_  256
#define H_  256
#define G4_ (4 * H_)   // 1024 fused gate width

typedef __attribute__((ext_vector_type(16))) __bf16 v16bf;
typedef __attribute__((ext_vector_type(8)))  __bf16 v8bf;
typedef __attribute__((ext_vector_type(8)))  float  v8f;

// ---------------------------------------------------------------------------
// WMMA helper: D(16x16,f32) += A(16x32,bf16) x B(32x16,bf16)
// ---------------------------------------------------------------------------
static __device__ inline v8f wmma_bf16(v16bf a, v16bf b, v8f c) {
  return __builtin_amdgcn_wmma_f32_16x16x32_bf16(
      /*neg_a=*/false, a, /*neg_b=*/false, b,
      /*c_mod=*/(short)0, c, /*reuse_a=*/false, /*reuse_b=*/false);
}

// A-fragment (16x32 bf16) from row-major bf16. rowp points at A[row] + kb*32.
// Layout: lanes 0-15: K = e; lanes 16-31: K = 16 + e (contiguous, matches packed B).
static __device__ inline v16bf load_a_frag(const __bf16* rowp, int g) {
  union { v8bf h[2]; v16bf v; } u;
  u.h[0] = *reinterpret_cast<const v8bf*>(rowp + g * 16);      // K = 16g .. 16g+7
  u.h[1] = *reinterpret_cast<const v8bf*>(rowp + g * 16 + 8);  // K = 16g+8 .. 16g+15
  return u.v;
}

// ---------------------------------------------------------------------------
// fp32 -> bf16 row-major copy
// ---------------------------------------------------------------------------
__global__ void f32_to_bf16_kernel(const float* __restrict__ in,
                                   __bf16* __restrict__ out, size_t n) {
  for (size_t i = (size_t)blockIdx.x * blockDim.x + threadIdx.x; i < n;
       i += (size_t)gridDim.x * blockDim.x)
    out[i] = (__bf16)in[i];
}

// ---------------------------------------------------------------------------
// Repack row-major fp32 W[K][N] into WMMA B-fragment stream:
// blocks (K/32)x(N/16), 512 bf16 per block; within a block lane l owns 16
// contiguous halves: N = ntile + (l&15), K = ktile + 16*(l>>4) + e.
// ---------------------------------------------------------------------------
__global__ void repack_b_kernel(const float* __restrict__ W,
                                __bf16* __restrict__ out, int K, int N) {
  size_t total = (size_t)K * N;
  int nblkN = N >> 4;
  for (size_t idx = (size_t)blockIdx.x * blockDim.x + threadIdx.x; idx < total;
       idx += (size_t)gridDim.x * blockDim.x) {
    size_t blk = idx >> 9;
    int within = (int)(idx & 511);
    int lane = within >> 4, e = within & 15;
    int nb = (int)(blk % nblkN), kb = (int)(blk / nblkN);
    int n = nb * 16 + (lane & 15);
    int k = kb * 32 + ((lane >> 4) << 4) + e;
    out[idx] = (__bf16)W[(size_t)k * N + n];
  }
}

// ---------------------------------------------------------------------------
// Zero LSTM initial state: h0 (bf16) and c (fp32), both [B,H]
// ---------------------------------------------------------------------------
__global__ void init_state_kernel(__bf16* __restrict__ h0,
                                  float* __restrict__ c, int n) {
  int i = blockIdx.x * blockDim.x + threadIdx.x;
  if (i < n) { h0[i] = (__bf16)0.0f; c[i] = 0.0f; }
}

// ---------------------------------------------------------------------------
// Generic GEMM: C[M,N] (fp32) = A[M,K] (bf16 row-major) x Bp (packed) + bias
// WG tile 128x256, 8 waves (2Mx4N), wave tile 64x64 (4 Mtiles x 4 Ntiles):
// 16 WMMAs per 8 fragment loads per K-step.
// ---------------------------------------------------------------------------
__global__ __launch_bounds__(256)
void gemm_bf16_wmma_kernel(const __bf16* __restrict__ A,
                           const __bf16* __restrict__ Bp,
                           const float* __restrict__ bias,
                           float* __restrict__ C, int M, int K, int N) {
  const int lane = threadIdx.x & 31;
  const int wave = threadIdx.x >> 5;
  const int g = lane >> 4;
  const int m15 = lane & 15;
  const int nWGn = N >> 8;                  // WGs along N (256-wide)
  const int wgM = blockIdx.x / nWGn;
  const int wgN = blockIdx.x % nWGn;
  const int baseM = wgM * 128 + (wave >> 2) * 64;
  const int baseN = wgN * 256 + (wave & 3) * 64;
  const int nbN = N >> 4;

  v8f acc[4][4];
#pragma unroll
  for (int i = 0; i < 4; ++i)
#pragma unroll
    for (int j = 0; j < 4; ++j) acc[i][j] = (v8f){0, 0, 0, 0, 0, 0, 0, 0};

  const int kBlocks = K >> 5;
  for (int kb = 0; kb < kBlocks; ++kb) {
    if (kb + 1 < kBlocks)  // prefetch next K-block of the packed B stream
      __builtin_prefetch(Bp + ((size_t)(kb + 1) * nbN + (baseN >> 4)) * 512 +
                             lane * 16, 0, 0);
    v16bf af[4];
#pragma unroll
    for (int i = 0; i < 4; ++i) {
      int row = baseM + i * 16 + m15;
      af[i] = load_a_frag(A + (size_t)row * K + kb * 32, g);
    }
    v16bf bf[4];
#pragma unroll
    for (int j = 0; j < 4; ++j) {
      int nb = (baseN >> 4) + j;
      bf[j] = *reinterpret_cast<const v16bf*>(
          Bp + ((size_t)kb * nbN + nb) * 512 + lane * 16);
    }
#pragma unroll
    for (int i = 0; i < 4; ++i)
#pragma unroll
      for (int j = 0; j < 4; ++j) acc[i][j] = wmma_bf16(af[i], bf[j], acc[i][j]);
  }

#pragma unroll
  for (int i = 0; i < 4; ++i)
#pragma unroll
    for (int j = 0; j < 4; ++j) {
      int col = baseN + j * 16 + m15;
      float badd = bias ? bias[col] : 0.0f;
#pragma unroll
      for (int r = 0; r < 8; ++r) {
        int row = baseM + i * 16 + r + 8 * g;
        C[(size_t)row * N + col] = acc[i][j][r] + badd;
      }
    }
}

// ---------------------------------------------------------------------------
// One LSTM timestep. Grid = 16 WGs, each owns 16 hidden columns (all 4 gate
// quadrants) x all 64 batch rows. h_prev (32 KB) is staged into LDS with
// gfx1250 async-to-LDS copies (ASYNCcnt), then A fragments come from LDS.
// gates = xg[:,t,:] + h_prev @ Wh (WMMA), fused sigmoid/tanh state update.
// ---------------------------------------------------------------------------
static __device__ inline float sigm(float x) { return 1.0f / (1.0f + __expf(-x)); }

__global__ __launch_bounds__(256)
void lstm_step_kernel(const __bf16* __restrict__ hprev,   // [B,H] bf16
                      __bf16* __restrict__ hnext,         // [B,H] bf16
                      float* __restrict__ cst,            // [B,H] fp32
                      const __bf16* __restrict__ WhB,     // packed K=256,N=1024
                      const float* __restrict__ xg,       // [B*T, 4H], row b*T+t
                      int t) {
  __shared__ __align__(16) __bf16 shA[B_ * H_];  // 32 KB staged h_prev
  __shared__ float gl[64][65];                   // gate staging, padded stride
  const int lane = threadIdx.x & 31;
  const int wave = threadIdx.x >> 5;
  const int g = lane >> 4;
  const int m15 = lane & 15;
  const int n0 = blockIdx.x * 16;  // hidden columns n0..n0+15

  // Coalesced async copy of h_prev into LDS: 2048 x 16B chunks, 8 per thread.
  {
    unsigned base_lds = (unsigned)(uintptr_t)(&shA[0]);
    uint64_t base_g = (uint64_t)hprev;
#pragma unroll
    for (int i = 0; i < 8; ++i) {
      unsigned off = ((unsigned)threadIdx.x + i * 256) * 16;  // byte offset
      unsigned laddr = base_lds + off;
      uint64_t gaddr = base_g + off;
      asm volatile("global_load_async_to_lds_b128 %0, %1, off"
                   :: "v"(laddr), "v"(gaddr) : "memory");
    }
    asm volatile("s_wait_asynccnt 0x0" ::: "memory");
  }
  __syncthreads();

#pragma unroll
  for (int j = 0; j < 2; ++j) {
    int tidx = wave * 2 + j;          // 16 tiles: 4 Mtiles x 4 quadrants
    int mt = tidx & 3;                // M tile (rows mt*16..)
    int q = tidx >> 2;                // gate quadrant (i,f,g,o)
    int nb = ((q * H_ + n0) >> 4);    // B n-tile in full 1024-wide gate matrix
    v8f acc = (v8f){0, 0, 0, 0, 0, 0, 0, 0};
#pragma unroll
    for (int kb = 0; kb < (H_ >> 5); ++kb) {
      int row = mt * 16 + m15;
      v16bf a = load_a_frag(shA + (size_t)row * H_ + kb * 32, g);  // ds_load
      v16bf b = *reinterpret_cast<const v16bf*>(
          WhB + ((size_t)kb * (G4_ >> 4) + nb) * 512 + lane * 16);
      acc = wmma_bf16(a, b, acc);
    }
#pragma unroll
    for (int r = 0; r < 8; ++r) gl[mt * 16 + r + 8 * g][q * 16 + m15] = acc[r];
  }
  __syncthreads();

  for (int it = threadIdx.x; it < 64 * 16; it += 256) {
    int m = it >> 4;              // batch row
    int nn = it & 15;
    int hcol = n0 + nn;
    const float* xr = xg + ((size_t)m * T_ + t) * G4_;
    float gi = gl[m][0 * 16 + nn] + xr[0 * H_ + hcol];
    float gf = gl[m][1 * 16 + nn] + xr[1 * H_ + hcol];
    float gg = gl[m][2 * 16 + nn] + xr[2 * H_ + hcol];
    float go = gl[m][3 * 16 + nn] + xr[3 * H_ + hcol];
    float cn = sigm(gf) * cst[m * H_ + hcol] + sigm(gi) * tanhf(gg);
    cst[m * H_ + hcol] = cn;
    hnext[m * H_ + hcol] = (__bf16)(sigm(go) * tanhf(cn));
  }
}

// ---------------------------------------------------------------------------
// Fused: q(t=0) projection, attention row 0, output projection, MLP head.
// One WG (256 threads) per batch. All tiny reductions; no WMMA needed here.
// h1 = hidden states slot-major: h(t,b,:) at (t*B + b)*H; k,v rows are t*B+b.
// ---------------------------------------------------------------------------
__global__ __launch_bounds__(256)
void attn_head_kernel(const __bf16* __restrict__ h1,
                      const float* __restrict__ kbuf,
                      const float* __restrict__ vbuf,
                      const float* __restrict__ Wq, const float* __restrict__ bq,
                      const float* __restrict__ Wo, const float* __restrict__ bo,
                      const float* __restrict__ W1, const float* __restrict__ b1,
                      const float* __restrict__ W2, const float* __restrict__ b2,
                      float* __restrict__ out) {
  const int b = blockIdx.x;
  const int tid = threadIdx.x;
  __shared__ float h0[H_], q0[H_], o0[H_], o2[H_];
  __shared__ float sc[T_];
  __shared__ float red[256];
  __shared__ float z[32];

  h0[tid] = (float)h1[(size_t)b * H_ + tid];  // h(t=0, b)
  __syncthreads();

  // q0 = h0 @ Wq + bq, pre-scaled by 1/sqrt(H)
  {
    float acc = bq[tid];
    for (int j = 0; j < H_; ++j) acc += h0[j] * Wq[(size_t)j * H_ + tid];
    q0[tid] = acc * (1.0f / 16.0f);
  }
  __syncthreads();

  // scores for all 512 keys (2 per thread)
  float lmax = -3.4e38f;
  for (int t = tid; t < T_; t += 256) {
    const float* kr = kbuf + ((size_t)t * B_ + b) * H_;
    float s = 0.0f;
    for (int j = 0; j < H_; ++j) s += q0[j] * kr[j];
    sc[t] = s;
    lmax = fmaxf(lmax, s);
  }
  red[tid] = lmax;
  __syncthreads();
  for (int s = 128; s > 0; s >>= 1) {
    if (tid < s) red[tid] = fmaxf(red[tid], red[tid + s]);
    __syncthreads();
  }
  float mx = red[0];
  __syncthreads();

  float lsum = 0.0f;
  for (int t = tid; t < T_; t += 256) {
    float p = __expf(sc[t] - mx);
    sc[t] = p;
    lsum += p;
  }
  red[tid] = lsum;
  __syncthreads();
  for (int s = 128; s > 0; s >>= 1) {
    if (tid < s) red[tid] += red[tid + s];
    __syncthreads();
  }
  float inv = 1.0f / red[0];
  __syncthreads();

  // o0 = softmax @ V   (thread = output dim, coalesced v reads)
  {
    float acc = 0.0f;
    for (int t = 0; t < T_; ++t) acc += sc[t] * vbuf[((size_t)t * B_ + b) * H_ + tid];
    o0[tid] = acc * inv;
  }
  __syncthreads();

  // o2 = o0 @ Wo + bo
  {
    float acc = bo[tid];
    for (int j = 0; j < H_; ++j) acc += o0[j] * Wo[(size_t)j * H_ + tid];
    o2[tid] = acc;
  }
  __syncthreads();

  if (tid < 32) {
    float acc = b1[tid];
    for (int d = 0; d < H_; ++d) acc += o2[d] * W1[(size_t)d * 32 + tid];
    z[tid] = fmaxf(acc, 0.0f);
  }
  __syncthreads();

  if (tid < 3) {
    float acc = b2[tid];
    for (int j = 0; j < 32; ++j) acc += z[j] * W2[(size_t)j * 3 + tid];
    out[b * 3 + tid] = acc;
  }
}

// ---------------------------------------------------------------------------
// Host orchestration
// ---------------------------------------------------------------------------
extern "C" void kernel_launch(void* const* d_in, const int* in_sizes, int n_in,
                              void* d_out, int out_size, void* d_ws, size_t ws_size,
                              hipStream_t stream) {
  (void)in_sizes; (void)n_in; (void)out_size; (void)ws_size;
  const float* x  = (const float*)d_in[0];
  const float* Wi = (const float*)d_in[1];
  const float* Wh = (const float*)d_in[2];
  const float* bb = (const float*)d_in[3];
  const float* Wq = (const float*)d_in[4];
  const float* bq = (const float*)d_in[5];
  const float* Wk = (const float*)d_in[6];
  const float* bk = (const float*)d_in[7];
  const float* Wv = (const float*)d_in[8];
  const float* bv = (const float*)d_in[9];
  const float* Wo = (const float*)d_in[10];
  const float* bo = (const float*)d_in[11];
  const float* W1 = (const float*)d_in[12];
  const float* b1 = (const float*)d_in[13];
  const float* W2 = (const float*)d_in[14];
  const float* b2 = (const float*)d_in[15];
  float* out = (float*)d_out;

  // Workspace carve-up (all offsets 256B-aligned)
  char* w = (char*)d_ws;
  size_t off = 0;
  auto carve = [&](size_t bytes) -> char* {
    char* p = w + off;
    off += (bytes + 255) & ~(size_t)255;
    return p;
  };
  __bf16* x_bf  = (__bf16*)carve((size_t)B_ * T_ * D_ * 2);         // 16 MB
  __bf16* WiB   = (__bf16*)carve((size_t)D_ * G4_ * 2);             // 512 KB
  __bf16* WhB   = (__bf16*)carve((size_t)H_ * G4_ * 2);             // 512 KB
  __bf16* WkB   = (__bf16*)carve((size_t)H_ * H_ * 2);              // 128 KB
  __bf16* WvB   = (__bf16*)carve((size_t)H_ * H_ * 2);              // 128 KB
  float*  xg    = (float*) carve((size_t)B_ * T_ * G4_ * 4);        // 128 MB
  __bf16* h_all = (__bf16*)carve((size_t)(T_ + 1) * B_ * H_ * 2);   // ~16.8 MB
  float*  cst   = (float*) carve((size_t)B_ * H_ * 4);              // 64 KB
  float*  kbuf  = (float*) carve((size_t)B_ * T_ * H_ * 4);         // 32 MB
  float*  vbuf  = (float*) carve((size_t)B_ * T_ * H_ * 4);         // 32 MB

  // 1) convert x to bf16; repack weights into WMMA B-fragment layout
  f32_to_bf16_kernel<<<2048, 256, 0, stream>>>(x, x_bf, (size_t)B_ * T_ * D_);
  repack_b_kernel<<<512, 256, 0, stream>>>(Wi, WiB, D_, G4_);
  repack_b_kernel<<<512, 256, 0, stream>>>(Wh, WhB, H_, G4_);
  repack_b_kernel<<<128, 256, 0, stream>>>(Wk, WkB, H_, H_);
  repack_b_kernel<<<128, 256, 0, stream>>>(Wv, WvB, H_, H_);
  init_state_kernel<<<(B_ * H_ + 255) / 256, 256, 0, stream>>>(h_all, cst, B_ * H_);

  // 2) xg = x @ Wi + b  (32768x1024, K=256)
  gemm_bf16_wmma_kernel<<<(B_ * T_ / 128) * (G4_ / 256), 256, 0, stream>>>(
      x_bf, WiB, bb, xg, B_ * T_, D_, G4_);

  // 3) sequential LSTM scan: 512 small WMMA step kernels (graph replay
  //    amortizes launch cost; per-step working set is L2-resident)
  for (int t = 0; t < T_; ++t) {
    lstm_step_kernel<<<16, 256, 0, stream>>>(
        h_all + (size_t)t * B_ * H_, h_all + (size_t)(t + 1) * B_ * H_,
        cst, WhB, xg, t);
  }

  // 4) k,v projections over all timesteps (rows = t*B + b), slot-major h
  const __bf16* h1 = h_all + (size_t)B_ * H_;  // h(t=0..T-1)
  gemm_bf16_wmma_kernel<<<(B_ * T_ / 128) * (H_ / 256), 256, 0, stream>>>(
      h1, WkB, bk, kbuf, B_ * T_, H_, H_);
  gemm_bf16_wmma_kernel<<<(B_ * T_ / 128) * (H_ / 256), 256, 0, stream>>>(
      h1, WvB, bv, vbuf, B_ * T_, H_, H_);

  // 5) fused q(t=0) + attention row 0 + Wo + MLP head -> out [64,3]
  attn_head_kernel<<<B_, 256, 0, stream>>>(h1, kbuf, vbuf, Wq, bq, Wo, bo,
                                           W1, b1, W2, b2, out);
}
